// HybridGCN_62912680952407
// MI455X (gfx1250) — compile-verified, matrix operand order checked by reading, not run on hardware
//
#include <hip/hip_runtime.h>
#include <hip/hip_bf16.h>

#define N_NODES 50000
#define N_EDGES 1000000
#define EP_EDGES 400000
#define HDIM 128
#define CDIM 768

typedef __attribute__((ext_vector_type(16))) __bf16 v16bf;
typedef __attribute__((ext_vector_type(8)))  __bf16 v8bf;
typedef __attribute__((ext_vector_type(8)))  float  v8f;

// ---------------- small elementwise kernels ----------------

__global__ void zero_f32(float* p, int n) {
  int i = blockIdx.x * 256 + threadIdx.x;
  if (i < n) p[i] = 0.0f;
}

__global__ void deg_scatter(const int* __restrict__ col, float* deg, int e) {
  int i = blockIdx.x * 256 + threadIdx.x;
  if (i < e) unsafeAtomicAdd(&deg[col[i]], 1.0f);
}

__global__ void dinv_kernel(float* deg, int n) {
  int i = blockIdx.x * 256 + threadIdx.x;
  if (i < n) {
    float d = deg[i];
    deg[i] = (d > 0.0f) ? rsqrtf(fmaxf(d, 1.0f)) : 0.0f;
  }
}

__global__ void norm_kernel(const int* __restrict__ row, const int* __restrict__ col,
                            const float* __restrict__ dinv, float* norm, int e) {
  int i = blockIdx.x * 256 + threadIdx.x;
  if (i < e) norm[i] = dinv[row[i]] * dinv[col[i]];
}

__global__ void bias_relu(const float* __restrict__ agg, const float* __restrict__ bias,
                          float* __restrict__ zout, int total) {
  int i = blockIdx.x * 256 + threadIdx.x;
  if (i < total) zout[i] = fmaxf(agg[i] + bias[i & (HDIM - 1)], 0.0f);
}

// agg[col[e]][j] += h[row[e]][j] * norm[e]   (one lane per (edge, feature))
__global__ void scatter_agg(const float* __restrict__ h, const int* __restrict__ row,
                            const int* __restrict__ col, const float* __restrict__ norm,
                            float* agg, int total) {
  int idx = blockIdx.x * 256 + threadIdx.x;
  if (idx >= total) return;
  int e = idx >> 7;          // /HDIM
  int j = idx & (HDIM - 1);
  float v = h[(size_t)row[e] * HDIM + j] * norm[e];
  unsafeAtomicAdd(&agg[(size_t)col[e] * HDIM + j], v);
}

// ---------------- weight packing into WMMA B-fragment order ----------------
// Packed layout: P[((ntile*(K/32) + kt)*32 + lane)*16 + j]
//   n = ntile*16 + (lane&15); halfk = (lane>>4)*8
//   j<8 : k = kt*32 + halfk + j ;  j>=8 : k = kt*32 + 16 + halfk + (j-8)
__global__ void pack_w(const float* __restrict__ B, __bf16* __restrict__ P,
                       int K, int Nc, int total) {
  int idx = blockIdx.x * 256 + threadIdx.x;
  if (idx >= total) return;
  int j    = idx & 15;
  int lane = (idx >> 4) & 31;
  int rest = idx >> 9;
  int kt    = rest % (K >> 5);
  int ntile = rest / (K >> 5);
  int n     = ntile * 16 + (lane & 15);
  int halfk = (lane >> 4) * 8;
  int k = kt * 32 + ((j < 8) ? (halfk + j) : (16 + halfk + (j - 8)));
  P[idx] = (__bf16)B[(size_t)k * Nc + n];
}

// ---------------- bf16 WMMA GEMM: Cout(MxN) = A(MxK,f32) @ Bpacked(KxN,bf16) ----------------
// epilogue: +bias (per col), optional relu, optional per-row mask
__global__ void __launch_bounds__(256)
gemm_wmma(const float* __restrict__ A, const __bf16* __restrict__ Bp,
          const float* __restrict__ bias, const float* __restrict__ rowmask,
          float* __restrict__ Cout, int M, int K, int Nc, int relu) {
  int lane = threadIdx.x & 31;
  int wid  = threadIdx.x >> 5;
  int ntiles = Nc >> 4;
  int tile = blockIdx.x * 8 + wid;
  if (tile >= (M >> 4) * ntiles) return;
  int mt = tile / ntiles, nt = tile % ntiles;
  int m0 = mt << 4, n0 = nt << 4;
  int l15 = lane & 15;
  int halfk = (lane >> 4) * 8;
  const float*  arow  = A + (size_t)(m0 + l15) * K;
  const __bf16* bbase = Bp + (size_t)nt * (K >> 5) * 512 + lane * 16;
  v8f acc = {};
  int ksteps = K >> 5;
  for (int kt = 0; kt < ksteps; ++kt) {
    int k0 = kt << 5;
    v8f a0 = *(const v8f*)(arow + k0 + halfk);
    v8f a1 = *(const v8f*)(arow + k0 + 16 + halfk);
    v16bf a;
#pragma unroll
    for (int i = 0; i < 8; ++i) { a[i] = (__bf16)a0[i]; a[8 + i] = (__bf16)a1[i]; }
    v16bf b = *(const v16bf*)(bbase + (size_t)kt * 512);
    acc = __builtin_amdgcn_wmma_f32_16x16x32_bf16(false, a, false, b, (short)0, acc,
                                                  false, false);
  }
  int colj = n0 + l15;
  float bi = bias ? bias[colj] : 0.0f;
#pragma unroll
  for (int r = 0; r < 8; ++r) {
    int m = m0 + (lane >> 4) * 8 + r;
    float v = acc[r] + bi;
    if (relu) v = fmaxf(v, 0.0f);
    if (rowmask) v *= rowmask[m];
    Cout[(size_t)m * Nc + colj] = v;
  }
}

// ---------------- fused decoder: 16 edges per block ----------------
// f = [z[s]*z[d] || c[s]*c[d]] (16x256 bf16 in LDS) -> WMMA @ dec_w1 (256x128)
// -> relu -> dot(dec_w2) -> scalar per edge
__global__ void __launch_bounds__(256)
decode_kernel(const int* __restrict__ edges, const float* __restrict__ z,
              const float* __restrict__ c, const __bf16* __restrict__ w1p,
              const float* __restrict__ b1, const float* __restrict__ w2,
              const float* __restrict__ b2, float* __restrict__ out) {
  __shared__ __bf16 fT[16][256];
  __shared__ float  hm[16][HDIM];
  __shared__ float  red[256];
  __shared__ int    ss[16], sd[16];

  int tid = threadIdx.x;
  int e0  = blockIdx.x * 16;
  if (tid < 16) {
    ss[tid] = edges[(size_t)(e0 + tid) * 2];
    sd[tid] = edges[(size_t)(e0 + tid) * 2 + 1];
  }
  __syncthreads();

  // build feature tile: e = it, j = tid (coalesced gathers along features)
#pragma unroll
  for (int it = 0; it < 16; ++it) {
    int s = ss[it], d = sd[it];
    int j = tid;
    float v;
    if (j < HDIM) v = z[(size_t)s * HDIM + j] * z[(size_t)d * HDIM + j];
    else          v = c[(size_t)s * HDIM + (j - HDIM)] * c[(size_t)d * HDIM + (j - HDIM)];
    fT[it][j] = (__bf16)v;
  }
  __syncthreads();

  // WMMA: wave wid owns output columns [wid*16, wid*16+16), K = 256 (8 steps)
  int lane = tid & 31, wid = tid >> 5;
  int l15 = lane & 15;
  int halfk = (lane >> 4) * 8;
  const __bf16* bbase = w1p + (size_t)wid * 8 * 512 + lane * 16;
  v8f acc = {};
#pragma unroll
  for (int kt = 0; kt < 8; ++kt) {
    int k0 = kt * 32;
    v8bf a0 = *(const v8bf*)(&fT[l15][k0 + halfk]);
    v8bf a1 = *(const v8bf*)(&fT[l15][k0 + 16 + halfk]);
    v16bf a;
#pragma unroll
    for (int i = 0; i < 8; ++i) { a[i] = a0[i]; a[8 + i] = a1[i]; }
    v16bf b = *(const v16bf*)(bbase + kt * 512);
    acc = __builtin_amdgcn_wmma_f32_16x16x32_bf16(false, a, false, b, (short)0, acc,
                                                  false, false);
  }
  int coln = wid * 16 + l15;
  float bi = b1[coln];
#pragma unroll
  for (int r = 0; r < 8; ++r) {
    int m = (lane >> 4) * 8 + r;
    hm[m][coln] = fmaxf(acc[r] + bi, 0.0f);
  }
  __syncthreads();

  // final GEMV: out[e] = hm[e] . w2 + b2
  int e = tid >> 4, l16 = tid & 15;
  float p = 0.0f;
#pragma unroll
  for (int j = l16; j < HDIM; j += 16) p += hm[e][j] * w2[j];
  red[tid] = p;
  __syncthreads();
  if (tid < 16) {
    float s = b2[0];
#pragma unroll
    for (int i = 0; i < 16; ++i) s += red[tid * 16 + i];
    out[e0 + tid] = s;
  }
}

// ---------------- launch ----------------

extern "C" void kernel_launch(void* const* d_in, const int* in_sizes, int n_in,
                              void* d_out, int out_size, void* d_ws, size_t ws_size,
                              hipStream_t stream) {
  const int*   edge_index = (const int*)d_in[0];
  const float* chemistry  = (const float*)d_in[1];
  const int*   pos_edge   = (const int*)d_in[2];
  const int*   neg_edge   = (const int*)d_in[3];
  const float* smiles     = (const float*)d_in[4];
  const float* node_emb   = (const float*)d_in[5];
  const float* conv_w     = (const float*)d_in[6];
  const float* conv_b     = (const float*)d_in[7];
  const float* chem_w     = (const float*)d_in[8];
  const float* chem_b     = (const float*)d_in[9];
  const float* dec_w1     = (const float*)d_in[10];
  const float* dec_b1     = (const float*)d_in[11];
  const float* dec_w2     = (const float*)d_in[12];
  const float* dec_b2     = (const float*)d_in[13];
  float* out = (float*)d_out;

  char* p = (char*)d_ws;
  auto carve = [&](size_t bytes) {
    char* r = p;
    p += (bytes + 255) & ~(size_t)255;
    return r;
  };
  float*  deg  = (float*)carve((size_t)N_NODES * 4);
  float*  norm = (float*)carve((size_t)N_EDGES * 4);
  float*  bufA = (float*)carve((size_t)N_NODES * HDIM * 4);
  float*  bufB = (float*)carve((size_t)N_NODES * HDIM * 4);
  float*  cbuf = (float*)carve((size_t)N_NODES * HDIM * 4);
  __bf16* w0p  = (__bf16*)carve((size_t)HDIM * HDIM * 2);
  __bf16* w1p  = (__bf16*)carve((size_t)HDIM * HDIM * 2);
  __bf16* wcp  = (__bf16*)carve((size_t)CDIM * HDIM * 2);
  __bf16* wdp  = (__bf16*)carve((size_t)2 * HDIM * HDIM * 2);

  const int* row = edge_index;
  const int* col = edge_index + N_EDGES;

  // degree -> dinv -> per-edge norm
  zero_f32<<<(N_NODES + 255) / 256, 256, 0, stream>>>(deg, N_NODES);
  deg_scatter<<<(N_EDGES + 255) / 256, 256, 0, stream>>>(col, deg, N_EDGES);
  dinv_kernel<<<(N_NODES + 255) / 256, 256, 0, stream>>>(deg, N_NODES);
  norm_kernel<<<(N_EDGES + 255) / 256, 256, 0, stream>>>(row, col, deg, norm, N_EDGES);

  // pack all weights into WMMA B-fragment order (bf16)
  pack_w<<<(HDIM * HDIM + 255) / 256, 256, 0, stream>>>(conv_w, w0p, HDIM, HDIM, HDIM * HDIM);
  pack_w<<<(HDIM * HDIM + 255) / 256, 256, 0, stream>>>(conv_w + HDIM * HDIM, w1p, HDIM, HDIM, HDIM * HDIM);
  pack_w<<<(CDIM * HDIM + 255) / 256, 256, 0, stream>>>(chem_w, wcp, CDIM, HDIM, CDIM * HDIM);
  pack_w<<<(2 * HDIM * HDIM + 255) / 256, 256, 0, stream>>>(dec_w1, wdp, 2 * HDIM, HDIM, 2 * HDIM * HDIM);

  const int tilesNH  = (N_NODES / 16) * (HDIM / 16);
  const int gemmGrid = (tilesNH + 7) / 8;
  const int totalNH  = N_NODES * HDIM;
  const int totalEH  = N_EDGES * HDIM;

  // GCN layer 0: h = emb @ W0 ; agg = scatter ; z1 = relu(agg + b0)
  gemm_wmma<<<gemmGrid, 256, 0, stream>>>(node_emb, w0p, nullptr, nullptr, bufA,
                                          N_NODES, HDIM, HDIM, 0);
  zero_f32<<<(totalNH + 255) / 256, 256, 0, stream>>>(bufB, totalNH);
  scatter_agg<<<(totalEH + 255) / 256, 256, 0, stream>>>(bufA, row, col, norm, bufB, totalEH);
  bias_relu<<<(totalNH + 255) / 256, 256, 0, stream>>>(bufB, conv_b, bufA, totalNH);

  // GCN layer 1: h2 = z1 @ W1 ; agg ; zfinal = relu(agg + b1)  -> bufB
  gemm_wmma<<<gemmGrid, 256, 0, stream>>>(bufA, w1p, nullptr, nullptr, bufB,
                                          N_NODES, HDIM, HDIM, 0);
  zero_f32<<<(totalNH + 255) / 256, 256, 0, stream>>>(bufA, totalNH);
  scatter_agg<<<(totalEH + 255) / 256, 256, 0, stream>>>(bufB, row, col, norm, bufA, totalEH);
  bias_relu<<<(totalNH + 255) / 256, 256, 0, stream>>>(bufA, conv_b + HDIM, bufB, totalNH);

  // chemistry branch: c = relu(chem @ Wc + bc) * smiles_mask  -> cbuf
  gemm_wmma<<<gemmGrid, 256, 0, stream>>>(chemistry, wcp, chem_b, smiles, cbuf,
                                          N_NODES, CDIM, HDIM, 1);

  // fused decode (pos then neg)
  decode_kernel<<<EP_EDGES / 16, 256, 0, stream>>>(pos_edge, bufB, cbuf, wdp, dec_b1,
                                                   dec_w2, dec_b2, out);
  decode_kernel<<<EP_EDGES / 16, 256, 0, stream>>>(neg_edge, bufB, cbuf, wdp, dec_b1,
                                                   dec_w2, dec_b2, out + EP_EDGES);
}
